// SwitchMLP_44504451121543
// MI455X (gfx1250) — compile-verified, matrix operand order checked by reading, not run on hardware
//
#include <hip/hip_runtime.h>
#include <hip/hip_bf16.h>

// ---------------------------------------------------------------------------
// SwitchMLP (top-2 MoE) for MI455X / gfx1250.
//   T=8192 tokens, H=2048, F=5632, E=8, TOP_K=2.
// bf16 WMMA (v_wmma_f32_16x16x32_bf16) + Tensor Data Mover tile feeds.
// Outputs (concatenated): out[T*H] f32, logits[T*E] f32.
// Workspace (~196 MB):
//   x_bf16  : T*H  bf16 (32 MB)     combine : T*E f32    lists : E*T i32
//   counts  : E i32 (padded)        act     : T*F bf16 (92 MB, per-expert reuse)
//   stageG/stageU/stageD : F*H bf16 each (23 MB each, per-expert reuse)
// Stream order serializes all per-expert reuse hazards.
// ---------------------------------------------------------------------------

typedef __attribute__((ext_vector_type(16))) __bf16 v16bf;
typedef __attribute__((ext_vector_type(8)))  __bf16 bf16x8;
typedef __attribute__((ext_vector_type(4)))  __bf16 bf16x4;
typedef __attribute__((ext_vector_type(8)))  float  v8f;
typedef __attribute__((ext_vector_type(4)))  unsigned tdm_g0_t;
typedef __attribute__((ext_vector_type(8)))  unsigned tdm_g1_t;

#define TT 8192
#define HH 2048
#define FF 5632
#define EE 8

static __device__ __forceinline__ __bf16 bf16_zero() {
  return __builtin_bit_cast(__bf16, (unsigned short)0);
}

static __device__ __forceinline__ bf16x8 bf16x8_zero() {
  bf16x8 z;
#pragma unroll
  for (int q = 0; q < 8; ++q) z[q] = bf16_zero();
  return z;
}

// ---------------------------------------------------------------------------
// Tensor Data Mover: 2-D tile load (global -> LDS), D# per ISA ch.8.
//   group0: count=1 | lds_addr | global_addr[56:0] | type=2
//   group1: data_size | tensor_dim0/1 | tile_dim0/1 | tensor_dim0_stride
// 2-operand form (VADDR2/3 = NULL) is valid for tensors up to 2-D.
// Issued once per wave (scalar op, EXEC ignored); tracked by TENSORcnt.
// ---------------------------------------------------------------------------
static __device__ __forceinline__ void tdm_load_2d(
    const void* gsrc, unsigned lds_byte, unsigned data_size_code,
    unsigned tensor_d0, unsigned tensor_d1, unsigned stride_d0,
    unsigned tile_d0, unsigned tile_d1) {
  unsigned long long ga = (unsigned long long)(size_t)gsrc;
  tdm_g0_t g0;
  g0[0] = 1u;                                                   // count=1, user
  g0[1] = lds_byte;                                             // lds_addr
  g0[2] = (unsigned)ga;                                         // ga[31:0]
  g0[3] = (unsigned)((ga >> 32) & 0x01FFFFFFull) | (2u << 30);  // ga[56:32]|type=2
  tdm_g1_t g1;
  g1[0] = (data_size_code & 3u) << 16;          // wg_mask=0, data_size
  g1[1] = (tensor_d0 & 0xFFFFu) << 16;          // tensor_dim0 @ bit48
  g1[2] = ((tensor_d0 >> 16) & 0xFFFFu) | ((tensor_d1 & 0xFFFFu) << 16);
  g1[3] = ((tensor_d1 >> 16) & 0xFFFFu) | ((tile_d0 & 0xFFFFu) << 16);
  g1[4] = (tile_d1 & 0xFFFFu);                  // tile_dim1, tile_dim2=0
  g1[5] = stride_d0;                            // tensor_dim0_stride[31:0]
  g1[6] = 0u;
  g1[7] = 0u;
  asm volatile("tensor_load_to_lds %0, %1" : : "s"(g0), "s"(g1) : "memory");
}

// low 32 bits of a generic pointer to LDS == LDS byte offset (ISA 10.2)
static __device__ __forceinline__ unsigned lds_byte_addr(const void* p) {
  return (unsigned)(size_t)p;
}

// 16x32 bf16 fragment from LDS tile (row stride 32): lane L = row L&15,
// K[kb..kb+8) and K[kb+16..kb+24), kb = 8*(L>=16).  -> ds_load_b128 pair
static __device__ __forceinline__ v16bf lds_load_frag(const __bf16* base, int lane) {
  const int row = lane & 15;
  const int kb  = (lane >> 4) << 3;
  bf16x8 lo = *(const bf16x8*)(base + row * 32 + kb);
  bf16x8 hi = *(const bf16x8*)(base + row * 32 + kb + 16);
  return __builtin_shufflevector(lo, hi, 0, 1, 2, 3, 4, 5, 6, 7,
                                 8, 9, 10, 11, 12, 13, 14, 15);
}

// ---------------------------------------------------------------------------
// 1) prep: x -> bf16, out <- residual, zero expert counts
// ---------------------------------------------------------------------------
__global__ __launch_bounds__(256) void prep_kernel(
    const float* __restrict__ x, const float* __restrict__ res,
    __bf16* __restrict__ xb, float* __restrict__ out_main,
    int* __restrict__ counts) {
  size_t i4 = (size_t)blockIdx.x * blockDim.x + threadIdx.x;
  size_t base = i4 * 4;
  if (base < (size_t)TT * HH) {
    float4 xv = *(const float4*)(x + base);
    float4 rv = *(const float4*)(res + base);
    bf16x4 b;
    b[0] = (__bf16)xv.x; b[1] = (__bf16)xv.y;
    b[2] = (__bf16)xv.z; b[3] = (__bf16)xv.w;
    *(bf16x4*)(xb + base) = b;
    *(float4*)(out_main + base) = rv;
  }
  if (i4 < EE) counts[i4] = 0;
}

// ---------------------------------------------------------------------------
// 1b) weight convert: fp32 -> bf16 (RNE), streamed once per expert per launch
// ---------------------------------------------------------------------------
__global__ __launch_bounds__(256) void convert_w_kernel(
    const float* __restrict__ src, __bf16* __restrict__ dst) {
  size_t i4 = (size_t)blockIdx.x * blockDim.x + threadIdx.x;
  size_t base = i4 * 4;
  float4 v = *(const float4*)(src + base);
  bf16x4 b;
  b[0] = (__bf16)v.x; b[1] = (__bf16)v.y;
  b[2] = (__bf16)v.z; b[3] = (__bf16)v.w;
  *(bf16x4*)(dst + base) = b;
}

// ---------------------------------------------------------------------------
// 2) router: logits, top-2 softmax combine weights, per-expert token lists
// ---------------------------------------------------------------------------
__global__ __launch_bounds__(256) void router_kernel(
    const float* __restrict__ x, const float* __restrict__ rw,
    const int* __restrict__ limitp,
    float* __restrict__ logits_out, float* __restrict__ combine,
    int* __restrict__ counts, int* __restrict__ lists) {
  const int lane = threadIdx.x & 31;
  const int wave = threadIdx.x >> 5;
  const int t = blockIdx.x * 8 + wave;
  const float* xr = x + (size_t)t * HH;

  float acc[EE];
#pragma unroll
  for (int q = 0; q < EE; ++q) acc[q] = 0.f;

  for (int k = lane * 4; k < HH; k += 32 * 4) {
    float4 xv = *(const float4*)(xr + k);
#pragma unroll
    for (int q = 0; q < EE; ++q) {
      float4 wv = *(const float4*)(rw + (size_t)q * HH + k);
      acc[q] += xv.x * wv.x + xv.y * wv.y + xv.z * wv.z + xv.w * wv.w;
    }
  }
#pragma unroll
  for (int q = 0; q < EE; ++q)
#pragma unroll
    for (int off = 16; off >= 1; off >>= 1)
      acc[q] += __shfl_xor(acc[q], off, 32);

  if (lane == 0) {
    const int lim = *limitp;
    const float NEG_INF = -__builtin_inff();
    float lg[EE];
#pragma unroll
    for (int q = 0; q < EE; ++q) {
      lg[q] = (q < lim) ? acc[q] : NEG_INF;
      logits_out[(size_t)t * EE + q] = lg[q];
    }
    int i0 = 0;
#pragma unroll
    for (int q = 1; q < EE; ++q) if (lg[q] > lg[i0]) i0 = q;
    int i1 = (i0 == 0) ? 1 : 0;
#pragma unroll
    for (int q = 0; q < EE; ++q)
      if (q != i0 && lg[q] > lg[i1]) i1 = q;

    float e1 = __expf(lg[i1] - lg[i0]);
    float inv = 1.f / (1.f + e1);
#pragma unroll
    for (int q = 0; q < EE; ++q) combine[(size_t)t * EE + q] = 0.f;
    combine[(size_t)t * EE + i0] = inv;
    combine[(size_t)t * EE + i1] = e1 * inv;

    int s0 = atomicAdd(&counts[i0], 1); lists[i0 * TT + s0] = t;
    int s1 = atomicAdd(&counts[i1], 1); lists[i1 * TT + s1] = t;
  }
}

// ---------------------------------------------------------------------------
// 3) gateup: act = silu(x gW^T) * (x uW^T) for routed tokens of expert e.
//    64(M)x128(N) tile, K-step 32. Gathered A: per-lane b128 loads.
//    Weight tiles: TDM tensor_load_to_lds, double-buffered; wave 0 issues
//    next slab's TDM before compute, s_wait_tensorcnt + barrier publishes.
// ---------------------------------------------------------------------------
__global__ __launch_bounds__(256) void gateup_kernel(
    const __bf16* __restrict__ xb, const __bf16* __restrict__ sG,
    const __bf16* __restrict__ sU,
    const int* __restrict__ counts, const int* __restrict__ lists,
    __bf16* __restrict__ act, int e) {
  __shared__ __align__(16) __bf16 ldsA[2][64 * 32];
  __shared__ __align__(16) __bf16 ldsG[2][128 * 32];
  __shared__ __align__(16) __bf16 ldsU[2][128 * 32];

  const int cnt = counts[e];
  const int mbase = blockIdx.y * 64;
  if (mbase >= cnt) return;
  const int nbase = blockIdx.x * 128;

  const int tid = threadIdx.x;
  const int lane = tid & 31;
  const int wave = tid >> 5;
  const int wm = wave & 1;
  const int wn = wave >> 1;

  const int arow = tid >> 2;
  const int acol = (tid & 3) * 8;
  long a_src = -1;
  { int g = mbase + arow; if (g < cnt) a_src = lists[e * TT + g]; }

  const __bf16* gBase = sG + (size_t)nbase * HH;
  const __bf16* uBase = sU + (size_t)nbase * HH;

  v8f accG[2][2], accU[2][2];
#pragma unroll
  for (int i = 0; i < 2; ++i)
#pragma unroll
    for (int j = 0; j < 2; ++j)
#pragma unroll
      for (int v = 0; v < 8; ++v) { accG[i][j][v] = 0.f; accU[i][j][v] = 0.f; }

  const int nK = HH / 32;

  // prologue: stage slab 0
  {
    bf16x8 av = bf16x8_zero();
    if (a_src >= 0) av = *(const bf16x8*)(xb + (size_t)a_src * HH + acol);
    *(bf16x8*)(&ldsA[0][arow * 32 + acol]) = av;
    if (wave == 0) {
      tdm_load_2d(gBase, lds_byte_addr(&ldsG[0][0]), 1u, HH, FF, HH, 32u, 128u);
      tdm_load_2d(uBase, lds_byte_addr(&ldsU[0][0]), 1u, HH, FF, HH, 32u, 128u);
    }
  }

  for (int ks = 0; ks < nK; ++ks) {
    const int cur = ks & 1, nxt = cur ^ 1;
    if (wave == 0) __builtin_amdgcn_s_wait_tensorcnt(0);  // slab cur landed
    __syncthreads();            // publish TDM + A stores; frees nxt buffers
    if (ks + 1 < nK) {
      if (wave == 0) {          // overlap next slab's DMA with compute
        tdm_load_2d(gBase + (ks + 1) * 32, lds_byte_addr(&ldsG[nxt][0]),
                    1u, HH, FF, HH, 32u, 128u);
        tdm_load_2d(uBase + (ks + 1) * 32, lds_byte_addr(&ldsU[nxt][0]),
                    1u, HH, FF, HH, 32u, 128u);
      }
      bf16x8 av = bf16x8_zero();
      if (a_src >= 0)
        av = *(const bf16x8*)(xb + (size_t)a_src * HH + (ks + 1) * 32 + acol);
      *(bf16x8*)(&ldsA[nxt][arow * 32 + acol]) = av;
    }

    v16bf af[2], bg[2], bu[2];
#pragma unroll
    for (int i = 0; i < 2; ++i)
      af[i] = lds_load_frag(&ldsA[cur][(wm * 32 + i * 16) * 32], lane);
#pragma unroll
    for (int j = 0; j < 2; ++j) {
      bg[j] = lds_load_frag(&ldsG[cur][(wn * 32 + j * 16) * 32], lane);
      bu[j] = lds_load_frag(&ldsU[cur][(wn * 32 + j * 16) * 32], lane);
    }
#pragma unroll
    for (int i = 0; i < 2; ++i)
#pragma unroll
      for (int j = 0; j < 2; ++j) {
        accG[i][j] = __builtin_amdgcn_wmma_f32_16x16x32_bf16(
            false, af[i], false, bg[j], (short)0, accG[i][j], false, false);
        accU[i][j] = __builtin_amdgcn_wmma_f32_16x16x32_bf16(
            false, af[i], false, bu[j], (short)0, accU[i][j], false, false);
      }
  }

  // fused SiLU(g) * u epilogue -> bf16 act
  const int coln = lane & 15;
  const int rsel = (lane >> 4) << 3;        // C/D layout: m = v + 8*(lane>=16)
#pragma unroll
  for (int i = 0; i < 2; ++i)
#pragma unroll
    for (int j = 0; j < 2; ++j)
#pragma unroll
      for (int v = 0; v < 8; ++v) {
        int gr = mbase + wm * 32 + i * 16 + rsel + v;
        if (gr < cnt) {
          float g = accG[i][j][v];
          float u = accU[i][j][v];
          float s = g * (1.f / (1.f + __expf(-g)));
          act[(size_t)gr * FF + nbase + wn * 32 + j * 16 + coln] = (__bf16)(s * u);
        }
      }
}

// ---------------------------------------------------------------------------
// 4) down: out[t,:] += combine[t,e] * (act @ dW^T + b). Both tiles via TDM
//    (act rows are compacted & contiguous; rows >= cnt read as zero via
//    tensor_dim OOB). Epilogue: relaxed agent-scope float atomics.
// ---------------------------------------------------------------------------
__global__ __launch_bounds__(256) void down_kernel(
    const __bf16* __restrict__ act, const __bf16* __restrict__ sD,
    const float* __restrict__ downB, const float* __restrict__ combine,
    const int* __restrict__ counts, const int* __restrict__ lists,
    float* __restrict__ out, int e) {
  __shared__ __align__(16) __bf16 ldsA[2][64 * 32];
  __shared__ __align__(16) __bf16 ldsB[2][128 * 32];

  const int cnt = counts[e];
  const int mbase = blockIdx.y * 64;
  if (mbase >= cnt) return;
  const int nbase = blockIdx.x * 128;

  const int tid = threadIdx.x;
  const int lane = tid & 31;
  const int wave = tid >> 5;
  const int wm = wave & 1;
  const int wn = wave >> 1;

  const __bf16* aBase = act + (size_t)mbase * FF;
  const __bf16* bBase = sD + (size_t)nbase * FF;
  const unsigned arows = (unsigned)(cnt - mbase);   // OOB rows -> zeros

  v8f acc[2][2];
#pragma unroll
  for (int i = 0; i < 2; ++i)
#pragma unroll
    for (int j = 0; j < 2; ++j)
#pragma unroll
      for (int v = 0; v < 8; ++v) acc[i][j][v] = 0.f;

  const int nK = FF / 32;
  if (wave == 0) {
    tdm_load_2d(aBase, lds_byte_addr(&ldsA[0][0]), 1u, FF, arows, FF, 32u, 64u);
    tdm_load_2d(bBase, lds_byte_addr(&ldsB[0][0]), 1u, FF, HH, FF, 32u, 128u);
  }

  for (int ks = 0; ks < nK; ++ks) {
    const int cur = ks & 1, nxt = cur ^ 1;
    if (wave == 0) __builtin_amdgcn_s_wait_tensorcnt(0);
    __syncthreads();
    if (ks + 1 < nK && wave == 0) {
      tdm_load_2d(aBase + (ks + 1) * 32, lds_byte_addr(&ldsA[nxt][0]),
                  1u, FF, arows, FF, 32u, 64u);
      tdm_load_2d(bBase + (ks + 1) * 32, lds_byte_addr(&ldsB[nxt][0]),
                  1u, FF, HH, FF, 32u, 128u);
    }

    v16bf af[2], bf[2];
#pragma unroll
    for (int i = 0; i < 2; ++i)
      af[i] = lds_load_frag(&ldsA[cur][(wm * 32 + i * 16) * 32], lane);
#pragma unroll
    for (int j = 0; j < 2; ++j)
      bf[j] = lds_load_frag(&ldsB[cur][(wn * 32 + j * 16) * 32], lane);
#pragma unroll
    for (int i = 0; i < 2; ++i)
#pragma unroll
      for (int j = 0; j < 2; ++j)
        acc[i][j] = __builtin_amdgcn_wmma_f32_16x16x32_bf16(
            false, af[i], false, bf[j], (short)0, acc[i][j], false, false);
  }

  const int coln = lane & 15;
  const int rsel = (lane >> 4) << 3;
#pragma unroll
  for (int i = 0; i < 2; ++i)
#pragma unroll
    for (int v = 0; v < 8; ++v) {
      int gr = mbase + wm * 32 + i * 16 + rsel + v;
      if (gr < cnt) {
        int t = lists[e * TT + gr];
        float w = combine[(size_t)t * EE + e];
#pragma unroll
        for (int j = 0; j < 2; ++j) {
          int col = nbase + wn * 32 + j * 16 + coln;
          float val = w * (acc[i][j][v] + downB[(size_t)e * HH + col]);
          __hip_atomic_fetch_add(out + (size_t)t * HH + col, val,
                                 __ATOMIC_RELAXED, __HIP_MEMORY_SCOPE_AGENT);
        }
      }
    }
}

// ---------------------------------------------------------------------------
extern "C" void kernel_launch(void* const* d_in, const int* in_sizes, int n_in,
                              void* d_out, int out_size, void* d_ws, size_t ws_size,
                              hipStream_t stream) {
  (void)in_sizes; (void)n_in; (void)out_size; (void)ws_size;
  const float* x   = (const float*)d_in[0];
  const float* res = (const float*)d_in[1];
  const float* rw  = (const float*)d_in[2];
  const float* gW  = (const float*)d_in[3];
  const float* uW  = (const float*)d_in[4];
  const float* dW  = (const float*)d_in[5];
  const float* dB  = (const float*)d_in[6];
  const int*   lim = (const int*)d_in[7];

  float* out_main = (float*)d_out;
  float* logits   = out_main + (size_t)TT * HH;

  char* ws = (char*)d_ws;
  __bf16* xb      = (__bf16*)ws;  ws += (size_t)TT * HH * 2;
  float*  combine = (float*)ws;   ws += (size_t)TT * EE * 4;
  int*    lists   = (int*)ws;     ws += (size_t)EE * TT * 4;
  int*    counts  = (int*)ws;     ws += 256;
  __bf16* act     = (__bf16*)ws;  ws += (size_t)TT * FF * 2;
  __bf16* stageG  = (__bf16*)ws;  ws += (size_t)FF * HH * 2;
  __bf16* stageU  = (__bf16*)ws;  ws += (size_t)FF * HH * 2;
  __bf16* stageD  = (__bf16*)ws;

  prep_kernel<<<(TT * HH) / (256 * 4), 256, 0, stream>>>(x, res, xb, out_main, counts);
  router_kernel<<<TT / 8, 256, 0, stream>>>(x, rw, lim, logits, combine, counts, lists);

  const int cvt_blocks = (FF * HH) / (256 * 4);   // exact: F*H % 1024 == 0
  for (int e = 0; e < EE; ++e) {
    convert_w_kernel<<<cvt_blocks, 256, 0, stream>>>(gW + (size_t)e * FF * HH, stageG);
    convert_w_kernel<<<cvt_blocks, 256, 0, stream>>>(uW + (size_t)e * FF * HH, stageU);
    convert_w_kernel<<<cvt_blocks, 256, 0, stream>>>(dW + (size_t)e * HH * FF, stageD);
    dim3 g1(FF / 128, TT / 64);
    gateup_kernel<<<g1, 256, 0, stream>>>(xb, stageG, stageU, counts, lists, act, e);
    dim3 g2(HH / 128, TT / 64);
    down_kernel<<<g2, 256, 0, stream>>>(act, stageD, dB, combine, counts, lists, out_main, e);
  }
}
